// GraphAttentionNetwork_5909874999358
// MI455X (gfx1250) — matrix operand order repaired and checked by reading, not compiled, not run on hardware
//
#include <hip/hip_runtime.h>
#include <hip/hip_fp16.h>

// ---------------------------------------------------------------------------
// Problem constants (from setup_inputs): B=16, N=256, Fin=128, E=16, Hid=256,
// heads=8, hs=32. Layer 2: heads=1, hs=1.
// ---------------------------------------------------------------------------
#define BB   16
#define NN   256
#define FIN  128
#define EE   16
#define HID  256
#define HH   8
#define HS   32
#define GN   4        // query nodes per attention block (async double-buffer)
#define EPAD 4        // pad edge LDS row to 20 floats (80B, 16B aligned)

typedef __attribute__((ext_vector_type(16))) _Float16 v16h;
typedef __attribute__((ext_vector_type(8)))  float    v8f;
typedef __attribute__((ext_vector_type(4)))  unsigned int u32x4;

union FragU { u32x4 u[2]; v16h h; };
union Half8U { _Float16 h[8]; u32x4 u; };

// ---------------------------------------------------------------------------
// WMMA fragment loaders (CDNA5 wave32 layouts).
// A 16x32 f16: lane L, row M=L&15, hi=L>>4; halves[0..7]=A[M][kb+8hi+0..7],
//              halves[8..15]=A[M][kb+16+8hi+0..7].
// B 32x16 f16: lane L, col=L&15, hi=L>>4; halves[0..7]=B[kb+8hi+0..7][col],
//              halves[8..15]=B[kb+16+8hi+0..7][col] (column-major: rows of W / hT).
// C/D f32 16x16: lane L, col=L&15; element v -> row M = v + 8*(L>>4).
// ---------------------------------------------------------------------------
__device__ __forceinline__ v16h load_frag_rowA(const _Float16* base, int strideH,
                                               int kb, int lane) {
  int row = lane & 15, hi = lane >> 4;
  const _Float16* p = base + (size_t)row * strideH + kb + 8 * hi;
  FragU f;
  f.u[0] = *reinterpret_cast<const u32x4*>(p);
  f.u[1] = *reinterpret_cast<const u32x4*>(p + 16);
  return f.h;
}

__device__ __forceinline__ v16h load_frag_colB(const _Float16* base, int strideH,
                                               int kb, int lane) {
  int col = lane & 15, hi = lane >> 4;
  const _Float16* p = base + (size_t)col * strideH + kb + 8 * hi;
  FragU f;
  f.u[0] = *reinterpret_cast<const u32x4*>(p);
  f.u[1] = *reinterpret_cast<const u32x4*>(p + 16);
  return f.h;
}

__device__ __forceinline__ v8f wmma16(v16h a, v16h b, v8f c) {
  return __builtin_amdgcn_wmma_f32_16x16x32_f16(false, a, false, b,
                                                (short)0, c, false, false);
}

// ---------------------------------------------------------------------------
// gfx1250 async global->LDS copy (ASYNCcnt path).  LDS destination address is
// the low 32 bits of the generic pointer (LDS aperture keeps the LDS offset
// in addr[31:0] per the CDNA5 aperture mapping).
// ---------------------------------------------------------------------------
__device__ __forceinline__ void async_copy_b128(const void* gaddr, void* lds) {
  unsigned ldsoff = (unsigned)(uintptr_t)lds;
  asm volatile("global_load_async_to_lds_b128 %0, %1, off"
               :: "v"(ldsoff), "v"(gaddr) : "memory");
}
__device__ __forceinline__ void wait_async0() {
  asm volatile("s_wait_asynccnt 0x0" ::: "memory");
}

__device__ __forceinline__ float wave_max(float v) {
#pragma unroll
  for (int m = 16; m > 0; m >>= 1) v = fmaxf(v, __shfl_xor(v, m, 32));
  return v;
}
__device__ __forceinline__ float wave_sum(float v) {
#pragma unroll
  for (int m = 16; m > 0; m >>= 1) v += __shfl_xor(v, m, 32);
  return v;
}

// ---------------------------------------------------------------------------
// K0: f32 -> f16 weight conversion
// ---------------------------------------------------------------------------
__global__ void f32_to_f16_kernel(const float* __restrict__ src,
                                  _Float16* __restrict__ dst, int n) {
  int i = blockIdx.x * blockDim.x + threadIdx.x;
  if (i < n) dst[i] = (_Float16)src[i];
}

// ---------------------------------------------------------------------------
// K1: h = LN(x) @ W^T + b   (per block: 16 rows x all 256 outputs, 8 waves)
// Writes h row-major f16 (B,N,HID) and transposed hT f16 (B,HID,N).
// ---------------------------------------------------------------------------
template <int F>
__global__ void ln_linear_kernel(const float* __restrict__ x,      // (B,N,F)
                                 const float* __restrict__ lns,
                                 const float* __restrict__ lnb,
                                 const _Float16* __restrict__ Wh,  // (HID,F)
                                 const float* __restrict__ bias,
                                 _Float16* __restrict__ h_row,     // (B,N,HID)
                                 _Float16* __restrict__ hT) {      // (B,HID,N)
  constexpr int PAD = 8;
  __shared__ float xs[16][F];
  __shared__ __attribute__((aligned(16))) _Float16 an[16][F + PAD];
  __shared__ float red1[16][16];
  __shared__ float red2[16][16];
  __shared__ float muS[16], rsS[16];

  const int blk = blockIdx.x;              // B * (N/16)
  const int b   = blk >> 4;
  const int nt  = (blk & 15) * 16;
  const int tid = threadIdx.x;

  const float* xb = x + ((size_t)b * NN + nt) * F;
  for (int i = tid; i < 16 * F; i += 256) xs[i / F][i % F] = xb[i];
  __syncthreads();

  const int r = tid >> 4, c = tid & 15;
  float s = 0.f, ss = 0.f;
  for (int k = c; k < F; k += 16) { float v = xs[r][k]; s += v; ss += v * v; }
  red1[r][c] = s; red2[r][c] = ss;
  __syncthreads();
  if (tid < 16) {
    float a = 0.f, q = 0.f;
    for (int j = 0; j < 16; ++j) { a += red1[tid][j]; q += red2[tid][j]; }
    float m = a / (float)F;
    muS[tid] = m;
    rsS[tid] = rsqrtf(q / (float)F - m * m + 1e-5f);
  }
  __syncthreads();

  for (int i = tid; i < 16 * F; i += 256) {
    int rr = i / F, k = i % F;
    float v = (xs[rr][k] - muS[rr]) * rsS[rr] * lns[k] + lnb[k];
    an[rr][k] = (_Float16)v;
  }
  __syncthreads();

  const int wv = tid >> 5, lane = tid & 31;
  const int ob0 = wv * 32, ob1 = ob0 + 16;
  v8f acc0 = {}, acc1 = {};
  for (int kb = 0; kb < F; kb += 32) {
    v16h a  = load_frag_rowA(&an[0][0], F + PAD, kb, lane);
    v16h b0 = load_frag_colB(Wh + (size_t)ob0 * F, F, kb, lane);
    v16h b1 = load_frag_colB(Wh + (size_t)ob1 * F, F, kb, lane);
    acc0 = wmma16(a, b0, acc0);
    acc1 = wmma16(a, b1, acc1);
  }

  const int col = lane & 15, hi = lane >> 4;
  const float bi0 = bias[ob0 + col], bi1 = bias[ob1 + col];
  _Float16* hrb = h_row + ((size_t)b * NN + nt) * HID;
  _Float16* hTb = hT + (size_t)b * HID * NN;
  Half8U t0, t1;
#pragma unroll
  for (int v = 0; v < 8; ++v) {
    int row = v + 8 * hi;
    float v0 = acc0[v] + bi0, v1 = acc1[v] + bi1;
    hrb[(size_t)row * HID + ob0 + col] = (_Float16)v0;
    hrb[(size_t)row * HID + ob1 + col] = (_Float16)v1;
    t0.h[v] = (_Float16)v0; t1.h[v] = (_Float16)v1;
  }
  *reinterpret_cast<u32x4*>(hTb + (size_t)(ob0 + col) * NN + nt + 8 * hi) = t0.u;
  *reinterpret_cast<u32x4*>(hTb + (size_t)(ob1 + col) * NN + nt + 8 * hi) = t1.u;
}

// ---------------------------------------------------------------------------
// K2a: src/dst attention terms. src[b][h][n] = h[b,n,h,:]·a_src[h], etc.
// ---------------------------------------------------------------------------
__global__ void srcdst_kernel(const _Float16* __restrict__ h_row,
                              const float* __restrict__ A,   // (HH, 80)
                              float* __restrict__ srcv,      // (B,HH,N)
                              float* __restrict__ dstv) {
  int p = blockIdx.x * blockDim.x + threadIdx.x;
  if (p >= BB * HH * NN) return;
  int n = p & (NN - 1), h = (p >> 8) & (HH - 1), b = p >> 11;
  const _Float16* hv = h_row + ((size_t)b * NN + n) * HID + h * HS;
  const float* as = A + h * (2 * HS + EE);
  const float* ad = as + HS;
  float s0 = 0.f, s1 = 0.f;
#pragma unroll
  for (int d = 0; d < HS; ++d) {
    float hh = (float)hv[d];
    s0 += hh * as[d]; s1 += hh * ad[d];
  }
  srcv[p] = s0; dstv[p] = s1;
}

// ---------------------------------------------------------------------------
// K2b: logits + masked softmax -> coeffs f16 (B,HH,N,N).
// One block per (b, group of GN query nodes). Edge tiles (16KB each) are
// double-buffered in LDS via global_load_async_to_lds_b128: tile g+1 is
// prefetched (ASYNCcnt) while softmax for tile g runs. Thread t owns
// neighbor m=t; after logits land in LDS, wave w owns head h=w and does
// shuffle-based max/sum reductions, then writes its coeff row contiguously.
// ---------------------------------------------------------------------------
__global__ void attn_softmax_kernel(const float* __restrict__ edge,
                                    const int* __restrict__ adj,
                                    const float* __restrict__ A,
                                    const float* __restrict__ srcv,
                                    const float* __restrict__ dstv,
                                    _Float16* __restrict__ coeffs) {
  __shared__ __attribute__((aligned(16))) float ebuf[2][NN][EE + EPAD]; // 40KB
  __shared__ float lgS[HH][NN + 1];                                     // ~8KB
  __shared__ float aeS[HH][EE];
  __shared__ float sS[HH];

  const int blk = blockIdx.x;          // B * (N/GN)
  const int b  = blk / (NN / GN);
  const int n0 = (blk % (NN / GN)) * GN;
  const int t  = threadIdx.x;          // == m
  const int wv = t >> 5, lane = t & 31;

  if (t < HH * EE) aeS[t >> 4][t & 15] = A[(t >> 4) * (2 * HS + EE) + 2 * HS + (t & 15)];

  // prefetch tile 0
  {
    const float* g0 = edge + (((size_t)b * NN + n0) * NN + t) * EE;
#pragma unroll
    for (int c2 = 0; c2 < 4; ++c2)
      async_copy_b128(g0 + c2 * 4, &ebuf[0][t][c2 * 4]);
  }

  for (int g = 0; g < GN; ++g) {
    const int n = n0 + g;
    const int sel = g & 1;
    wait_async0();                       // this wave's tile-g loads done
    if (t < HH) sS[t] = srcv[((size_t)b * HH + t) * NN + n];
    __syncthreads();                     // whole tile visible, sS ready

    if (g + 1 < GN) {                    // prefetch next tile into other buffer
      const float* gn = edge + (((size_t)b * NN + n + 1) * NN + t) * EE;
#pragma unroll
      for (int c2 = 0; c2 < 4; ++c2)
        async_copy_b128(gn + c2 * 4, &ebuf[sel ^ 1][t][c2 * 4]);
    }

    // per-thread logits for all heads from LDS edge row
    float e[EE];
    const float* er = &ebuf[sel][t][0];
#pragma unroll
    for (int i = 0; i < EE; ++i) e[i] = er[i];
    const int av = adj[((size_t)b * NN + n) * NN + t];
#pragma unroll
    for (int h = 0; h < HH; ++h) {
      float acc = sS[h] + dstv[((size_t)b * HH + h) * NN + t];
#pragma unroll
      for (int i = 0; i < EE; ++i) acc += e[i] * aeS[h][i];
      lgS[h][t] = av ? acc : -1e30f;
    }
    __syncthreads();

    // wave w reduces head h=w: max, exp-sum, normalized write
    float vals[NN / 32];
    float vmax = -1e30f;
#pragma unroll
    for (int j = 0; j < NN / 32; ++j) {
      vals[j] = lgS[wv][lane + 32 * j];
      vmax = fmaxf(vmax, vals[j]);
    }
    vmax = wave_max(vmax);
    float vsum = 0.f;
#pragma unroll
    for (int j = 0; j < NN / 32; ++j) {
      float ex = __expf(vals[j] - vmax);
      vals[j] = ex;
      vsum += ex;
    }
    vsum = wave_sum(vsum);
    const float inv = 1.f / vsum;
    _Float16* cb = coeffs + (((size_t)b * HH + wv) * NN + n) * NN;
#pragma unroll
    for (int j = 0; j < NN / 32; ++j)
      cb[lane + 32 * j] = (_Float16)(vals[j] * inv);
    __syncthreads();                     // lgS / ebuf reuse guard
  }
}

// ---------------------------------------------------------------------------
// K3: aggregation GEMM. agg[b][n][h*32+d] = sum_m coeffs[b][h][n][m]*hT[b][h*32+d][m]
// One wave per (b, h, n-tile16, d-tile16); A/B fragments directly from global.
// ---------------------------------------------------------------------------
__global__ void aggregate_kernel(const _Float16* __restrict__ coeffs,
                                 const _Float16* __restrict__ hT,
                                 _Float16* __restrict__ agg) {
  const int gw = (blockIdx.x * blockDim.x + threadIdx.x) >> 5;  // 0..4095
  const int lane = threadIdx.x & 31;
  const int dt  = gw & 1;
  const int ntl = (gw >> 1) & 15;
  const int h   = (gw >> 5) & 7;
  const int b   = gw >> 8;

  const _Float16* cA = coeffs + (((size_t)b * HH + h) * NN + ntl * 16) * NN;
  const _Float16* bB = hT + ((size_t)b * HID + h * HS + dt * 16) * NN;

  v8f acc = {};
  for (int mb = 0; mb < NN; mb += 32) {
    v16h a = load_frag_rowA(cA, NN, mb, lane);
    v16h bf = load_frag_colB(bB, NN, mb, lane);
    acc = wmma16(a, bf, acc);
  }
  const int col = lane & 15, hi = lane >> 4;
  _Float16* ob = agg + (((size_t)b * NN + ntl * 16) * HID) + h * HS + dt * 16 + col;
#pragma unroll
  for (int v = 0; v < 8; ++v) ob[(size_t)(v + 8 * hi) * HID] = (_Float16)acc[v];
}

// ---------------------------------------------------------------------------
// K4: out = leaky_relu(LN(agg @ Wo^T + bo)) -> next layer f32 input
// ---------------------------------------------------------------------------
__global__ void out_ln_kernel(const _Float16* __restrict__ agg,  // (B,N,HID) f16
                              const _Float16* __restrict__ Woh,  // (HID,HID) f16
                              const float* __restrict__ bo,
                              const float* __restrict__ lns,
                              const float* __restrict__ lnb,
                              float* __restrict__ xout) {        // (B,N,HID) f32
  __shared__ float outS[16][HID + 4];
  __shared__ float red1[16][16], red2[16][16];
  __shared__ float muS[16], rsS[16];

  const int blk = blockIdx.x;
  const int b = blk >> 4, nt = (blk & 15) * 16;
  const int tid = threadIdx.x, wv = tid >> 5, lane = tid & 31;

  const _Float16* aA = agg + ((size_t)b * NN + nt) * HID;
  const int ob0 = wv * 32, ob1 = ob0 + 16;
  v8f acc0 = {}, acc1 = {};
  for (int kb = 0; kb < HID; kb += 32) {
    v16h a  = load_frag_rowA(aA, HID, kb, lane);
    v16h b0 = load_frag_colB(Woh + (size_t)ob0 * HID, HID, kb, lane);
    v16h b1 = load_frag_colB(Woh + (size_t)ob1 * HID, HID, kb, lane);
    acc0 = wmma16(a, b0, acc0);
    acc1 = wmma16(a, b1, acc1);
  }
  const int col = lane & 15, hi = lane >> 4;
  const float bi0 = bo[ob0 + col], bi1 = bo[ob1 + col];
#pragma unroll
  for (int v = 0; v < 8; ++v) {
    int row = v + 8 * hi;
    outS[row][ob0 + col] = acc0[v] + bi0;
    outS[row][ob1 + col] = acc1[v] + bi1;
  }
  __syncthreads();

  const int r = tid >> 4, c = tid & 15;
  float s = 0.f, ss = 0.f;
  for (int k = c; k < HID; k += 16) { float v = outS[r][k]; s += v; ss += v * v; }
  red1[r][c] = s; red2[r][c] = ss;
  __syncthreads();
  if (tid < 16) {
    float a = 0.f, q = 0.f;
    for (int j = 0; j < 16; ++j) { a += red1[tid][j]; q += red2[tid][j]; }
    float m = a / (float)HID;
    muS[tid] = m;
    rsS[tid] = rsqrtf(q / (float)HID - m * m + 1e-5f);
  }
  __syncthreads();

  float* xo = xout + ((size_t)b * NN + nt) * HID;
  for (int k = c; k < HID; k += 16) {
    float v = (outS[r][k] - muS[r]) * rsS[r] * lns[k] + lnb[k];
    v = v > 0.f ? v : 0.2f * v;
    xo[(size_t)r * HID + k] = v;
  }
}

// ---------------------------------------------------------------------------
// Final layer (heads=1, hs=1): tiny scalar pipeline.
// ---------------------------------------------------------------------------
__global__ void final_h_kernel(const float* __restrict__ x,     // (B,N,HID)
                               const float* __restrict__ lns,
                               const float* __restrict__ lnb,
                               const float* __restrict__ W2,    // (1,HID)
                               const float* __restrict__ b2,
                               float* __restrict__ h2) {        // (B,N)
  __shared__ float xs[16][HID];
  __shared__ float red1[16][16], red2[16][16];
  __shared__ float muS[16], rsS[16];
  const int blk = blockIdx.x;
  const int b = blk >> 4, nt = (blk & 15) * 16;
  const int tid = threadIdx.x;

  const float* xb = x + ((size_t)b * NN + nt) * HID;
  for (int i = tid; i < 16 * HID; i += 256) xs[i >> 8][i & 255] = xb[i];
  __syncthreads();
  const int r = tid >> 4, c = tid & 15;
  float s = 0.f, ss = 0.f;
  for (int k = c; k < HID; k += 16) { float v = xs[r][k]; s += v; ss += v * v; }
  red1[r][c] = s; red2[r][c] = ss;
  __syncthreads();
  if (tid < 16) {
    float a = 0.f, q = 0.f;
    for (int j = 0; j < 16; ++j) { a += red1[tid][j]; q += red2[tid][j]; }
    float m = a / (float)HID;
    muS[tid] = m;
    rsS[tid] = rsqrtf(q / (float)HID - m * m + 1e-5f);
  }
  __syncthreads();
  float dp = 0.f;
  for (int k = c; k < HID; k += 16) {
    float v = (xs[r][k] - muS[r]) * rsS[r] * lns[k] + lnb[k];
    dp += v * W2[k];
  }
  red1[r][c] = dp;
  __syncthreads();
  if (tid < 16) {
    float a = 0.f;
    for (int j = 0; j < 16; ++j) a += red1[tid][j];
    h2[(size_t)b * NN + nt + tid] = a + b2[0];
  }
}

__global__ void final_attn_kernel(const float* __restrict__ edge,
                                  const int* __restrict__ adj,
                                  const float* __restrict__ A2,  // (1,18)
                                  const float* __restrict__ h2,  // (B,N)
                                  const float* __restrict__ Wo2,
                                  const float* __restrict__ bo2,
                                  float* __restrict__ per_n) {   // (B,N)
  __shared__ float red[256];
  __shared__ float aeS[EE];
  const int blk = blockIdx.x;
  const int b = blk >> 8, n = blk & (NN - 1);
  const int t = threadIdx.x;  // == m
  if (t < EE) aeS[t] = A2[2 + t];
  __syncthreads();

  const float hm = h2[(size_t)b * NN + t];
  const float hn = h2[(size_t)b * NN + n];
  const float* ep = edge + (((size_t)b * NN + n) * NN + t) * EE;
  float acc = A2[0] * hn + A2[1] * hm;
#pragma unroll
  for (int i = 0; i < EE; ++i) acc += ep[i] * aeS[i];
  const int av = adj[((size_t)b * NN + n) * NN + t];
  float lg = av ? acc : -1e30f;

  red[t] = lg; __syncthreads();
  for (int s2 = 128; s2 > 0; s2 >>= 1) { if (t < s2) red[t] = fmaxf(red[t], red[t + s2]); __syncthreads(); }
  float mx = red[0]; __syncthreads();
  float ex = __expf(lg - mx);
  red[t] = ex; __syncthreads();
  for (int s2 = 128; s2 > 0; s2 >>= 1) { if (t < s2) red[t] += red[t + s2]; __syncthreads(); }
  float sm = red[0]; __syncthreads();
  red[t] = (ex / sm) * hm; __syncthreads();
  for (int s2 = 128; s2 > 0; s2 >>= 1) { if (t < s2) red[t] += red[t + s2]; __syncthreads(); }
  if (t == 0) per_n[(size_t)b * NN + n] = red[0] * Wo2[0] + bo2[0];
}

__global__ void final_reduce_kernel(const float* __restrict__ per_n,
                                    float* __restrict__ out) {
  __shared__ float red[256];
  const int b = blockIdx.x, t = threadIdx.x;
  red[t] = per_n[(size_t)b * NN + t];
  __syncthreads();
  for (int s2 = 128; s2 > 0; s2 >>= 1) { if (t < s2) red[t] += red[t + s2]; __syncthreads(); }
  if (t == 0) out[b] = red[0];
}

// ---------------------------------------------------------------------------
// Host side
// ---------------------------------------------------------------------------
extern "C" void kernel_launch(void* const* d_in, const int* in_sizes, int n_in,
                              void* d_out, int out_size, void* d_ws, size_t ws_size,
                              hipStream_t stream) {
  (void)in_sizes; (void)n_in; (void)out_size; (void)ws_size;
  const float* node = (const float*)d_in[0];
  const float* edge = (const float*)d_in[1];
  const int*   adj  = (const int*)d_in[2];
  const float* ln1s0 = (const float*)d_in[3];
  const float* ln1b0 = (const float*)d_in[4];
  const float* W0    = (const float*)d_in[5];
  const float* b0    = (const float*)d_in[6];
  const float* A0    = (const float*)d_in[7];
  const float* Wo0   = (const float*)d_in[8];
  const float* bo0   = (const float*)d_in[9];
  const float* ln2s0 = (const float*)d_in[10];
  const float* ln2b0 = (const float*)d_in[11];
  const float* ln1s1 = (const float*)d_in[12];
  const float* ln1b1 = (const float*)d_in[13];
  const float* W1    = (const float*)d_in[14];
  const float* b1    = (const float*)d_in[15];
  const float* A1    = (const float*)d_in[16];
  const float* Wo1   = (const float*)d_in[17];
  const float* bo1   = (const float*)d_in[18];
  const float* ln2s1 = (const float*)d_in[19];
  const float* ln2b1 = (const float*)d_in[20];
  const float* ln1s2 = (const float*)d_in[21];
  const float* ln1b2 = (const float*)d_in[22];
  const float* W2    = (const float*)d_in[23];
  const float* b2    = (const float*)d_in[24];
  const float* A2    = (const float*)d_in[25];
  const float* Wo2   = (const float*)d_in[26];
  const float* bo2   = (const float*)d_in[27];

  char* ws = (char*)d_ws;
  size_t off = 0;
  auto alloc = [&](size_t bytes) -> void* {
    void* p = ws + off;
    off = (off + bytes + 255) & ~(size_t)255;
    return p;
  };

  _Float16* wW0  = (_Float16*)alloc((size_t)HID * FIN * 2);
  _Float16* wW1  = (_Float16*)alloc((size_t)HID * HID * 2);
  _Float16* wWo0 = (_Float16*)alloc((size_t)HID * HID * 2);
  _Float16* wWo1 = (_Float16*)alloc((size_t)HID * HID * 2);
  _Float16* h_row = (_Float16*)alloc((size_t)BB * NN * HID * 2);
  _Float16* hT    = (_Float16*)alloc((size_t)BB * HID * NN * 2);
  float* srcv = (float*)alloc((size_t)BB * HH * NN * 4);
  float* dstv = (float*)alloc((size_t)BB * HH * NN * 4);
  _Float16* coeffs = (_Float16*)alloc((size_t)BB * HH * NN * NN * 2);
  _Float16* agg    = (_Float16*)alloc((size_t)BB * NN * HID * 2);
  float* x1 = (float*)alloc((size_t)BB * NN * HID * 4);
  float* x2 = (float*)alloc((size_t)BB * NN * HID * 4);
  float* h2 = (float*)alloc((size_t)BB * NN * 4);
  float* per_n = (float*)alloc((size_t)BB * NN * 4);

  f32_to_f16_kernel<<<(HID * FIN + 255) / 256, 256, 0, stream>>>(W0, wW0, HID * FIN);
  f32_to_f16_kernel<<<(HID * HID + 255) / 256, 256, 0, stream>>>(W1, wW1, HID * HID);
  f32_to_f16_kernel<<<(HID * HID + 255) / 256, 256, 0, stream>>>(Wo0, wWo0, HID * HID);
  f32_to_f16_kernel<<<(HID * HID + 255) / 256, 256, 0, stream>>>(Wo1, wWo1, HID * HID);

  const int gridLN   = BB * (NN / 16);        // 256
  const int gridSD   = (BB * HH * NN) / 256;  // 128
  const int gridAttn = BB * (NN / GN);        // 1024
  const int gridFin  = BB * NN;               // 4096
  const int gridAgg  = (BB * HH * (NN / 16) * 2) / 8;  // 512 blocks of 8 waves

  // ---- Layer 0 ----
  ln_linear_kernel<FIN><<<gridLN, 256, 0, stream>>>(node, ln1s0, ln1b0, wW0, b0, h_row, hT);
  srcdst_kernel<<<gridSD, 256, 0, stream>>>(h_row, A0, srcv, dstv);
  attn_softmax_kernel<<<gridAttn, 256, 0, stream>>>(edge, adj, A0, srcv, dstv, coeffs);
  aggregate_kernel<<<gridAgg, 256, 0, stream>>>(coeffs, hT, agg);
  out_ln_kernel<<<gridLN, 256, 0, stream>>>(agg, wWo0, bo0, ln2s0, ln2b0, x1);

  // ---- Layer 1 ----
  ln_linear_kernel<HID><<<gridLN, 256, 0, stream>>>(x1, ln1s1, ln1b1, wW1, b1, h_row, hT);
  srcdst_kernel<<<gridSD, 256, 0, stream>>>(h_row, A1, srcv, dstv);
  attn_softmax_kernel<<<gridAttn, 256, 0, stream>>>(edge, adj, A1, srcv, dstv, coeffs);
  aggregate_kernel<<<gridAgg, 256, 0, stream>>>(coeffs, hT, agg);
  out_ln_kernel<<<gridLN, 256, 0, stream>>>(agg, wWo1, bo1, ln2s1, ln2b1, x2);

  // ---- Layer 2 (1 head, scalar) ----
  final_h_kernel<<<gridLN, 256, 0, stream>>>(x2, ln1s2, ln1b2, W2, b2, h2);
  final_attn_kernel<<<gridFin, 256, 0, stream>>>(edge, adj, A2, h2, Wo2, bo2, per_n);
  final_reduce_kernel<<<BB, 256, 0, stream>>>(per_n, (float*)d_out);
}
